// LSTM_42013370090067
// MI455X (gfx1250) — matrix operand order repaired and checked by reading, not compiled, not run on hardware
//
#include <hip/hip_runtime.h>
#include <hip/hip_bf16.h>

// ---------------------------------------------------------------------------
// LSTM (H=512, T=16384, batch=1) for MI455X / gfx1250, wave32.
//   K1: x_gates = x @ W_ih^T + (b_ih+b_hh)   -> WMMA f16 GEMM (bandwidth bound)
//   K2: persistent 16-WGP recurrence. W_hh staged into 320KB LDS by the
//       Tensor Data Mover (TENSOR_LOAD_TO_LDS, TENSORcnt) with hardware
//       bank-padding; cluster/atomic barrier per step; h double-buffered in L2.
//   K3: inter = hs @ fc_w^T + fc_b           -> WMMA f16 GEMM
//   K4: fused heads fc1/fc2 + clamp
// ---------------------------------------------------------------------------

typedef __attribute__((ext_vector_type(16))) _Float16 v16h;
typedef __attribute__((ext_vector_type(8)))  _Float16 v8h;
typedef __attribute__((ext_vector_type(8)))  float    v8f;
typedef __attribute__((ext_vector_type(4)))  unsigned tdm_v4u;
typedef __attribute__((ext_vector_type(8)))  int      tdm_v8i;
typedef __attribute__((ext_vector_type(4)))  int      tdm_v4i;

#define HDIM 512
#define G4   2048          // 4*H
#define FCN  128
#define NWG  16
#define CLAMP_MAX_F 49688.0f

// ---------------- f32 -> f16 conversion ----------------
__global__ void cvt_f32_f16(const float* __restrict__ src,
                            _Float16* __restrict__ dst, int n) {
  int i = blockIdx.x * blockDim.x + threadIdx.x;
  if (i < n) dst[i] = (_Float16)src[i];
}

// ---------------- reset h double-buffer + grid barrier counter -------------
__global__ void reset_state(float* hbuf, unsigned* counter) {
  int i = threadIdx.x;
  if (i < 2 * HDIM) hbuf[i] = 0.0f;
  if (i == 0) *counter = 0u;
}

// ---------------- K1: x_gates WMMA GEMM ------------------------------------
// A = x_f16 [T,512] row-major, B[k][n] = W_ih[n][k] (W_ih_f16 [2048,512] rm).
__global__ __launch_bounds__(256) void xgates_gemm(
    const _Float16* __restrict__ A, const _Float16* __restrict__ W,
    const float* __restrict__ b0, const float* __restrict__ b1,
    float* __restrict__ C, int T) {
  const int lane   = threadIdx.x & 31;
  const int wave   = threadIdx.x >> 5;
  const int mTile  = blockIdx.y;
  const int nTile  = blockIdx.x * 8 + wave;
  const int l16    = lane & 15;
  const int hiHalf = lane >> 4;                       // K-half select per ISA layout
  const int m = mTile * 16 + l16;
  const int n = nTile * 16 + l16;
  const _Float16* Arow = A + (size_t)m * HDIM + hiHalf * 8;   // K {0..7,16..23}/{8..15,24..31}
  const _Float16* Brow = W + (size_t)n * HDIM + hiHalf * 8;   // K {0..7,16..23}/{8..15,24..31}
  v8f acc = {};
#pragma unroll 4
  for (int k0 = 0; k0 < HDIM; k0 += 32) {
    v8h a_lo = *(const v8h*)(Arow + k0);
    v8h a_hi = *(const v8h*)(Arow + k0 + 16);
    v8h b_lo = *(const v8h*)(Brow + k0);
    v8h b_hi = *(const v8h*)(Brow + k0 + 16);
    v16h a = __builtin_shufflevector(a_lo, a_hi, 0,1,2,3,4,5,6,7,8,9,10,11,12,13,14,15);
    v16h b = __builtin_shufflevector(b_lo, b_hi, 0,1,2,3,4,5,6,7,8,9,10,11,12,13,14,15);
    acc = __builtin_amdgcn_wmma_f32_16x16x32_f16(false, a, false, b,
                                                 (short)0, acc, false, false);
  }
  const float bias = b0[n] + b1[n];
  float* Crow = C + (size_t)(mTile * 16 + hiHalf * 8) * G4 + n;
#pragma unroll
  for (int r = 0; r < 8; ++r) Crow[(size_t)r * G4] = acc[r] + bias;
}

// ---------------- K3: inter WMMA GEMM --------------------------------------
__global__ __launch_bounds__(256) void inter_gemm(
    const _Float16* __restrict__ A, const _Float16* __restrict__ W,
    const float* __restrict__ bias, float* __restrict__ C, int T) {
  const int lane   = threadIdx.x & 31;
  const int wave   = threadIdx.x >> 5;          // nTile = wave (128/16 = 8 tiles)
  const int mTile  = blockIdx.y;
  const int l16    = lane & 15;
  const int hiHalf = lane >> 4;
  const int m = mTile * 16 + l16;
  const int n = wave * 16 + l16;
  const _Float16* Arow = A + (size_t)m * HDIM + hiHalf * 8;
  const _Float16* Brow = W + (size_t)n * HDIM + hiHalf * 8;
  v8f acc = {};
#pragma unroll 4
  for (int k0 = 0; k0 < HDIM; k0 += 32) {
    v8h a_lo = *(const v8h*)(Arow + k0);
    v8h a_hi = *(const v8h*)(Arow + k0 + 16);
    v8h b_lo = *(const v8h*)(Brow + k0);
    v8h b_hi = *(const v8h*)(Brow + k0 + 16);
    v16h a = __builtin_shufflevector(a_lo, a_hi, 0,1,2,3,4,5,6,7,8,9,10,11,12,13,14,15);
    v16h b = __builtin_shufflevector(b_lo, b_hi, 0,1,2,3,4,5,6,7,8,9,10,11,12,13,14,15);
    acc = __builtin_amdgcn_wmma_f32_16x16x32_f16(false, a, false, b,
                                                 (short)0, acc, false, false);
  }
  const float bb = bias[n];
  float* Crow = C + (size_t)(mTile * 16 + hiHalf * 8) * FCN + n;
#pragma unroll
  for (int r = 0; r < 8; ++r) Crow[(size_t)r * FCN] = acc[r] + bb;
}

// ---------------- K2: persistent LSTM recurrence ---------------------------
// 16 WGPs; WGP w owns h slice [w*32, w*32+32): gate rows (g*512 + w*32 + j).
// W_hh staged by TDM with pad_enable: 4 DWORD pad every 256 DWORDs
//   -> LDS row stride 520 f32, K-half offset 260 f32 (16B aligned,
//      64-bank perfect float4 reads: banks 4*(2r+half)+j, 2 req/bank).
#define WROW_STRIDE 520
#define WHALF_OFF   260
__device__ __forceinline__ float sigm(float x) { return 1.0f / (1.0f + __expf(-x)); }

__global__ __launch_bounds__(256) void lstm_rec(
    const float* __restrict__ xg, const float* __restrict__ Whh,
    float* __restrict__ hbuf, unsigned* __restrict__ counter,
    _Float16* __restrict__ hs, int T) {
  extern __shared__ float lds[];                 // dynamic LDS starts at offset 0
  float* Wl = lds;                               // 128 rows * 520 f32
  float* hl = lds + 128 * WROW_STRIDE;           // 512
  float* gb = hl + HDIM;                         // 128 gate pre-activations
  const int w   = blockIdx.x;
  const int tid = threadIdx.x;

  // --- TDM: DMA 4 gate blocks (32 rows x 512 f32 each) of W_hh into LDS ---
  if ((tid >> 5) == 0) {                         // wave 0 issues the descriptors
#pragma unroll
    for (int g = 0; g < 4; ++g) {
      const unsigned long long ga =
          (unsigned long long)(const void*)(Whh + (size_t)(g * HDIM + w * 32) * HDIM);
      const unsigned ldsOff = (unsigned)(g * 32 * WROW_STRIDE * 4);  // bytes
      // D# group 0: count=1 | lds_addr | global_addr[56:0] | type=2
      tdm_v4u g0;
      g0[0] = 1u;                                            // count=1, user mode
      g0[1] = ldsOff;                                        // lds_addr (bytes)
      g0[2] = (unsigned)(ga & 0xFFFFFFFFu);                  // global_addr[31:0]
      g0[3] = (unsigned)((ga >> 32) & 0x1FFFFFFu) | (2u << 30);  // [56:32] | type=2
      // D# group 1: data_size=4B, pad_enable, pad_interval=256dw, pad_amount=4dw,
      //             tensor_dim0=512, tensor_dim1=32, tile 512x32, dim0_stride=512
      tdm_v8i g1;
      g1[0] = (int)((2u << 16) | (1u << 20) | (7u << 22) | (3u << 25));
      g1[1] = (int)(((unsigned)HDIM & 0xFFFFu) << 16);       // tensor_dim0[15:0]
      g1[2] = (int)((32u & 0xFFFFu) << 16);                  // dim0[31:16]=0, tensor_dim1[15:0]=32
      g1[3] = (int)(((unsigned)HDIM) << 16);                 // tensor_dim1[31:16]=0, tile_dim0=512
      g1[4] = (int)32u;                                      // tile_dim1=32, tile_dim2=0
      g1[5] = (int)HDIM;                                     // tensor_dim0_stride[31:0]=512
      g1[6] = 0;                                             // stride[47:32], dim1_stride[15:0]
      g1[7] = 0;
      tdm_v4i g2 = {0, 0, 0, 0};                             // 2D tensor: groups 2/3 unused
      tdm_v4i g3 = {0, 0, 0, 0};
      tdm_v8i gx = {0, 0, 0, 0, 0, 0, 0, 0};                 // 6-arg toolchain extra group
      __builtin_amdgcn_tensor_load_to_lds(g0, g1, g2, g3, gx, 0);
    }
    __builtin_amdgcn_s_wait_tensorcnt(0);        // wave0's TENSORcnt -> 0
  }
  __syncthreads();                               // W_hh now LDS-resident

  const int r     = tid >> 1;                    // local gate row 0..127
  const int half  = tid & 1;                     // K half (0:[0,256) 1:[256,512))
  const int kbase = half * 256;                  // h-vector offset (unpadded)
  const int gr    = (r >> 5) * HDIM + w * 32 + (r & 31);
  const float* Wrow = Wl + r * WROW_STRIDE + (half ? WHALF_OFF : 0);
  float c = 0.0f;                                // cell state (threads 0..31)

  for (int t = 0; t < T; ++t) {
    // Stage h_{t} (written by all WGPs last step) from L2 into LDS.
    const float* hsrc = hbuf + (t & 1) * HDIM;
    for (int i = tid; i < HDIM; i += 256) hl[i] = hsrc[i];
    if (half == 0 && t + 1 < T)                  // prefetch next step's x_gates
      __builtin_prefetch(xg + (size_t)(t + 1) * G4 + gr, 0, 3);
    __syncthreads();

    // 256-long partial dot from LDS (bank-perfect float4 W reads, h broadcast).
    float acc = 0.0f;
#pragma unroll 8
    for (int j = 0; j < 256; j += 4) {
      float4 wv = *(const float4*)(Wrow + j);
      float4 hv = *(const float4*)(hl + kbase + j);
      acc = fmaf(wv.x, hv.x, acc);
      acc = fmaf(wv.y, hv.y, acc);
      acc = fmaf(wv.z, hv.z, acc);
      acc = fmaf(wv.w, hv.w, acc);
    }
    acc += __shfl_xor(acc, 1, 32);               // combine the two K-halves
    if (half == 0) gb[r] = acc + xg[(size_t)t * G4 + gr];  // xg includes biases
    __syncthreads();

    if (tid < 32) {                              // gate math for 32 h elements
      float gi = sigm(gb[tid]);
      float gf = sigm(gb[32 + tid]);
      float gg = tanhf(gb[64 + tid]);
      float go = sigm(gb[96 + tid]);
      c = gf * c + gi * gg;
      float h = go * tanhf(c);
      int hi = w * 32 + tid;
      hbuf[((t + 1) & 1) * HDIM + hi] = h;       // double-buffered h exchange
      hs[(size_t)t * HDIM + hi] = (_Float16)h;
    }

    // Inter-WGP step barrier: atomic sense + CDNA5 cluster barrier fast path
    __threadfence();
    __syncthreads();
    if (tid == 0) {
      atomicAdd(counter, 1u);
      unsigned target = (unsigned)(NWG * (t + 1));
      while (__hip_atomic_load(counter, __ATOMIC_RELAXED,
                               __HIP_MEMORY_SCOPE_AGENT) < target) { }
    }
    __builtin_amdgcn_s_cluster_barrier();        // NOP when unclustered (ISA §6.6)
    __syncthreads();
    __threadfence();
  }
}

// ---------------- K4: fused heads + clamp ----------------------------------
__global__ void heads(const float* __restrict__ inter,
                      const float* __restrict__ w1, const float* __restrict__ b1,
                      const float* __restrict__ w2, const float* __restrict__ b2,
                      float* __restrict__ out, int T) {
  int idx = blockIdx.x * blockDim.x + threadIdx.x;
  int total = T * 53;
  if (idx >= total) return;
  int row = idx / 53;
  int o   = idx - row * 53;
  const float* irow = inter + (size_t)row * FCN;
  const float* wrow;
  float acc;
  float* dst;
  if (o < 16) { wrow = w1 + o * FCN; acc = b1[o]; dst = out + (size_t)row * 16 + o; }
  else { int o2 = o - 16; wrow = w2 + o2 * FCN; acc = b2[o2];
         dst = out + (size_t)T * 16 + (size_t)row * 37 + o2; }
#pragma unroll 8
  for (int k = 0; k < FCN; k += 4) {
    float4 a = *(const float4*)(irow + k);
    float4 v = *(const float4*)(wrow + k);
    acc = fmaf(a.x, v.x, acc); acc = fmaf(a.y, v.y, acc);
    acc = fmaf(a.z, v.z, acc); acc = fmaf(a.w, v.w, acc);
  }
  *dst = fminf(fmaxf(acc, 0.0f), CLAMP_MAX_F);
}

// ---------------------------------------------------------------------------
extern "C" void kernel_launch(void* const* d_in, const int* in_sizes, int n_in,
                              void* d_out, int out_size, void* d_ws, size_t ws_size,
                              hipStream_t stream) {
  const float* x     = (const float*)d_in[0];
  const float* W_ih  = (const float*)d_in[1];
  const float* W_hh  = (const float*)d_in[2];
  const float* b_ih  = (const float*)d_in[3];
  const float* b_hh  = (const float*)d_in[4];
  const float* fc_w  = (const float*)d_in[5];
  const float* fc_b  = (const float*)d_in[6];
  const float* fc1_w = (const float*)d_in[7];
  const float* fc1_b = (const float*)d_in[8];
  const float* fc2_w = (const float*)d_in[9];
  const float* fc2_b = (const float*)d_in[10];
  const int T = in_sizes[0] / HDIM;              // 16384

  char* ws = (char*)d_ws;
  size_t off = 0;
  auto alloc = [&](size_t bytes) {
    char* p = ws + off;
    off = (off + bytes + 255) & ~(size_t)255;
    return p;
  };
  float*    xg      = (float*)   alloc((size_t)T * G4 * 4);     // 128 MB
  _Float16* x_h     = (_Float16*)alloc((size_t)T * HDIM * 2);   // 16 MB
  _Float16* Wih_h   = (_Float16*)alloc((size_t)G4 * HDIM * 2);  // 2 MB
  _Float16* fcw_h   = (_Float16*)alloc((size_t)FCN * HDIM * 2); // 128 KB
  _Float16* hs_h    = (_Float16*)alloc((size_t)T * HDIM * 2);   // 16 MB
  float*    inter   = (float*)   alloc((size_t)T * FCN * 4);    // 8 MB
  float*    hbuf    = (float*)   alloc(2 * HDIM * 4);
  unsigned* counter = (unsigned*)alloc(256);

  const int nx = T * HDIM;
  cvt_f32_f16<<<(nx + 255) / 256, 256, 0, stream>>>(x, x_h, nx);
  cvt_f32_f16<<<(G4 * HDIM + 255) / 256, 256, 0, stream>>>(W_ih, Wih_h, G4 * HDIM);
  cvt_f32_f16<<<(FCN * HDIM + 255) / 256, 256, 0, stream>>>(fc_w, fcw_h, FCN * HDIM);
  reset_state<<<1, 1024, 0, stream>>>(hbuf, counter);

  dim3 gA(G4 / 16 / 8, T / 16);                  // (16, 1024)
  xgates_gemm<<<gA, 256, 0, stream>>>(x_h, Wih_h, b_ih, b_hh, xg, T);

  const size_t ldsB = (size_t)(128 * WROW_STRIDE + HDIM + 128) * sizeof(float); // ~263 KB
  (void)hipFuncSetAttribute(reinterpret_cast<const void*>(&lstm_rec),
                            hipFuncAttributeMaxDynamicSharedMemorySize, (int)ldsB);
  lstm_rec<<<NWG, 256, ldsB, stream>>>(xg, W_hh, hbuf, counter, hs_h, T);

  inter_gemm<<<dim3(1, T / 16), 256, 0, stream>>>(hs_h, fcw_h, fc_b, inter, T);

  const int total = T * 53;
  heads<<<(total + 255) / 256, 256, 0, stream>>>(inter, fc1_w, fc1_b, fc2_w, fc2_b,
                                                 (float*)d_out, T);
}
